// MolecularGraphEncoder_9904194585206
// MI455X (gfx1250) — matrix operand order repaired and checked by reading, not compiled, not run on hardware
//
#include <hip/hip_runtime.h>

typedef __attribute__((ext_vector_type(16))) __bf16    v16bf;
typedef __attribute__((ext_vector_type(8)))  __bf16    v8bf;
typedef __attribute__((ext_vector_type(8)))  float     v8f;

#define N_ATOMS 3072
#define N_BONDS 12288
#define NGRAPH  64
#define DMODEL  256
#define NHEAD   8
#define NLAYER  4
#define HDIM    32

static __device__ __forceinline__ unsigned pk2(float a, float b)
{
    return ((unsigned)__builtin_bit_cast(unsigned short, (__bf16)b) << 16) |
           (unsigned)__builtin_bit_cast(unsigned short, (__bf16)a);
}

static __device__ __forceinline__ v16bf cat8(v8bf lo, v8bf hi)
{
    return __builtin_shufflevector(lo, hi, 0, 1, 2, 3, 4, 5, 6, 7,
                                           8, 9, 10, 11, 12, 13, 14, 15);
}

// ---------------------------------------------------------------------------
// bf16 GEMM: C[M,N] = act(A[M,K] @ W[N,K]^T + bias[N]); M%32==0, N%64==0,
// K%32==0. One wave computes a 32x64 tile (2x4 of 16x16x32 bf16 WMMAs).
// Epilogue can emit f32 C, bf16 C, and/or a bf16 transposed copy of the
// columns >= 512 (the V part of a QKV projection) as vT[col-512][row].
// ---------------------------------------------------------------------------
__global__ void gemm_bf16_wmma(const __bf16* __restrict__ A,
                               const __bf16* __restrict__ W,
                               const float* __restrict__ bias,
                               float* __restrict__ Cf,
                               __bf16* __restrict__ Cb,
                               __bf16* __restrict__ vT,
                               int M, int N, int K, int relu)
{
    const int lane = threadIdx.x & 31;
    const int wave = threadIdx.x >> 5;
    const int tilesN = N >> 6;
    const int tilesM = M >> 5;
    const int tile = blockIdx.x * 8 + wave;
    if (tile >= tilesM * tilesN) return;
    const int tm = (tile / tilesN) * 32;
    const int tn = (tile % tilesN) * 64;
    const int hf  = lane >> 4;
    const int l16 = lane & 15;

    v8f acc[2][4] = {};

    const __bf16* arow0 = A + (size_t)(tm + l16) * K;
    const __bf16* arow1 = A + (size_t)(tm + 16 + l16) * K;
    const __bf16* wrow[4];
#pragma unroll
    for (int j = 0; j < 4; ++j) wrow[j] = W + (size_t)(tn + j * 16 + l16) * K;

    for (int k0 = 0; k0 < K; k0 += 32) {
        // A fragments: 16 contiguous elems per lane at 16*hf (matches B)
        const v16bf a0 = cat8(*(const v8bf*)(arow0 + k0 + hf * 16),
                              *(const v8bf*)(arow0 + k0 + hf * 16 + 8));
        const v16bf a1 = cat8(*(const v8bf*)(arow1 + k0 + hf * 16),
                              *(const v8bf*)(arow1 + k0 + hf * 16 + 8));
        v16bf bfr[4];
#pragma unroll
        for (int j = 0; j < 4; ++j) {
            const __bf16* p = wrow[j] + k0 + hf * 16;     // 16 contiguous elems
            bfr[j] = cat8(*(const v8bf*)p, *(const v8bf*)(p + 8));
        }
#pragma unroll
        for (int j = 0; j < 4; ++j) {
            acc[0][j] = __builtin_amdgcn_wmma_f32_16x16x32_bf16(
                false, a0, false, bfr[j], (short)0, acc[0][j], false, false);
            acc[1][j] = __builtin_amdgcn_wmma_f32_16x16x32_bf16(
                false, a1, false, bfr[j], (short)0, acc[1][j], false, false);
        }
    }

    // epilogue: C layout — VGPR r: row = r + 8*hf, col = l16
#pragma unroll
    for (int mi = 0; mi < 2; ++mi) {
#pragma unroll
        for (int j = 0; j < 4; ++j) {
            const int col = tn + j * 16 + l16;
            const float b = bias[col];
#pragma unroll
            for (int r = 0; r < 8; ++r) {
                const int row = tm + mi * 16 + r + 8 * hf;
                float x = acc[mi][j][r] + b;
                if (relu) x = x > 0.f ? x : 0.f;
                if (Cf) Cf[(size_t)row * N + col] = x;
                if (Cb) Cb[(size_t)row * N + col] = (__bf16)x;
                if (vT && col >= 512) vT[(size_t)(col - 512) * M + row] = (__bf16)x;
            }
        }
    }
}

// ---------------------------------------------------------------------------
// Flash attention over bf16 QKV (+ transposed V). One wave = 16 queries of
// one head; keys streamed in chunks of 32. All fragments are 2x b128 loads.
// ---------------------------------------------------------------------------
__global__ void attn_flash(const __bf16* __restrict__ qkvb,  // [nq, 3*D] bf16
                           const __bf16* __restrict__ vT,    // [D, nq] bf16
                           const int* __restrict__ batch_idx,
                           __bf16* __restrict__ outp,        // [nq, D] bf16
                           int nq, int use_mask)
{
    __shared__ __align__(16) __bf16 plds[8][16 * 32];

    const int lane = threadIdx.x & 31;
    const int wave = threadIdx.x >> 5;
    const int task = blockIdx.x * 8 + wave;
    const int ntask = (nq / 16) * NHEAD;
    if (task >= ntask) return;
    const int h  = task % NHEAD;
    const int q0 = (task / NHEAD) * 16;
    const int hf  = lane >> 4;
    const int l16 = lane & 15;

    // Q fragment (A layout)
    const __bf16* qrow = qkvb + (size_t)(q0 + l16) * (3 * DMODEL) + h * HDIM;
    const v16bf qfrag = cat8(*(const v8bf*)(qrow + hf * 16),
                             *(const v8bf*)(qrow + hf * 16 + 8));

    // V^T row pointers (B layout for P@V: 16 contiguous keys per lane)
    const __bf16* vrow0 = vT + (size_t)(h * HDIM + l16) * nq;
    const __bf16* vrow1 = vT + (size_t)(h * HDIM + 16 + l16) * nq;

    int qb[8];
    if (use_mask) {
#pragma unroll
        for (int r = 0; r < 8; ++r) qb[r] = batch_idx[q0 + r + 8 * hf];
    }

    float mrow[8], lrow[8];
    v8f o0 = {}, o1 = {};
#pragma unroll
    for (int r = 0; r < 8; ++r) { mrow[r] = -1e30f; lrow[r] = 0.f; }

    const float scale = 0.17677669529663687f;  // 1/sqrt(32)

    for (int k0 = 0; k0 < nq; k0 += 32) {
        // K^T fragments (B layout): 16 contiguous hd elems at 16*hf
        const __bf16* kr0 = qkvb + (size_t)(k0 + l16)      * (3 * DMODEL) + DMODEL + h * HDIM + hf * 16;
        const __bf16* kr1 = qkvb + (size_t)(k0 + 16 + l16) * (3 * DMODEL) + DMODEL + h * HDIM + hf * 16;
        const v16bf kf0 = cat8(*(const v8bf*)kr0, *(const v8bf*)(kr0 + 8));
        const v16bf kf1 = cat8(*(const v8bf*)kr1, *(const v8bf*)(kr1 + 8));

        const v8f zero = {};
        v8f s0 = __builtin_amdgcn_wmma_f32_16x16x32_bf16(false, qfrag, false, kf0, (short)0, zero, false, false);
        v8f s1 = __builtin_amdgcn_wmma_f32_16x16x32_bf16(false, qfrag, false, kf1, (short)0, zero, false, false);

        int kb0 = 0, kb1 = 0;
        if (use_mask) { kb0 = batch_idx[k0 + l16]; kb1 = batch_idx[k0 + 16 + l16]; }

        float p0[8], p1[8];
#pragma unroll
        for (int r = 0; r < 8; ++r) {
            float a = s0[r] * scale;
            float b = s1[r] * scale;
            if (use_mask) {
                if (qb[r] != kb0) a = -1e9f;
                if (qb[r] != kb1) b = -1e9f;
            }
            float mx = fmaxf(a, b);
            mx = fmaxf(mx, __shfl_xor(mx, 1, 32));
            mx = fmaxf(mx, __shfl_xor(mx, 2, 32));
            mx = fmaxf(mx, __shfl_xor(mx, 4, 32));
            mx = fmaxf(mx, __shfl_xor(mx, 8, 32));
            const float nm = fmaxf(mrow[r], mx);
            const float corr = __expf(mrow[r] - nm);
            mrow[r] = nm;
            lrow[r] *= corr;
            o0[r] *= corr;
            o1[r] *= corr;
            p0[r] = __expf(a - nm);
            p1[r] = __expf(b - nm);
            float ps = p0[r] + p1[r];
            ps += __shfl_xor(ps, 1, 32);
            ps += __shfl_xor(ps, 2, 32);
            ps += __shfl_xor(ps, 4, 32);
            ps += __shfl_xor(ps, 8, 32);
            lrow[r] += ps;
        }

        // P: C layout -> LDS -> A layout (two 16B ds loads)
#pragma unroll
        for (int r = 0; r < 8; ++r) {
            plds[wave][(r + 8 * hf) * 32 + l16]      = (__bf16)p0[r];
            plds[wave][(r + 8 * hf) * 32 + 16 + l16] = (__bf16)p1[r];
        }
        asm volatile("s_wait_dscnt 0" ::: "memory");
        const v16bf pfrag = cat8(*(const v8bf*)&plds[wave][l16 * 32 + hf * 16],
                                 *(const v8bf*)&plds[wave][l16 * 32 + hf * 16 + 8]);
        asm volatile("s_wait_dscnt 0" ::: "memory");

        // V fragments from V^T: 16 contiguous keys per lane
        const v16bf vf0 = cat8(*(const v8bf*)(vrow0 + k0 + hf * 16),
                               *(const v8bf*)(vrow0 + k0 + hf * 16 + 8));
        const v16bf vf1 = cat8(*(const v8bf*)(vrow1 + k0 + hf * 16),
                               *(const v8bf*)(vrow1 + k0 + hf * 16 + 8));

        o0 = __builtin_amdgcn_wmma_f32_16x16x32_bf16(false, pfrag, false, vf0, (short)0, o0, false, false);
        o1 = __builtin_amdgcn_wmma_f32_16x16x32_bf16(false, pfrag, false, vf1, (short)0, o1, false, false);
    }

#pragma unroll
    for (int r = 0; r < 8; ++r) {
        const int q = q0 + r + 8 * hf;
        const float inv = 1.0f / lrow[r];
        outp[(size_t)q * DMODEL + h * HDIM + l16]      = (__bf16)(o0[r] * inv);
        outp[(size_t)q * DMODEL + h * HDIM + 16 + l16] = (__bf16)(o1[r] * inv);
    }
}

// ---------------------------------------------------------------------------
// Helpers
// ---------------------------------------------------------------------------
__global__ void zero_f32(float* __restrict__ p, int n)
{
    const int i = blockIdx.x * blockDim.x + threadIdx.x;
    if (i < n) p[i] = 0.f;
}

// f32 -> bf16, n even, packed u32 stores
__global__ void cvt_bf16(const float* __restrict__ s, __bf16* __restrict__ d, int n)
{
    const int i = blockIdx.x * blockDim.x + threadIdx.x;
    if (i >= n / 2) return;
    ((unsigned*)d)[i] = pk2(s[2 * i], s[2 * i + 1]);
}

// zero-pad rows: f32 [rows, kin] -> bf16 [rows, kout]
__global__ void pad_rows_bf(const float* __restrict__ src, __bf16* __restrict__ dst,
                            int rows, int kin, int kout)
{
    const int i = blockIdx.x * blockDim.x + threadIdx.x;
    if (i >= rows * kout) return;
    const int r = i / kout, c = i % kout;
    dst[i] = (c < kin) ? (__bf16)src[(size_t)r * kin + c] : (__bf16)0.f;
}

// messages (bf16) from f32 atom_h / bond_h
__global__ void build_messages(const float* __restrict__ atom_h,
                               const float* __restrict__ bond_h,
                               const int* __restrict__ src,
                               const int* __restrict__ tgt,
                               __bf16* __restrict__ msg)
{
    const int i = blockIdx.x * blockDim.x + threadIdx.x;   // float4 granularity
    if (i >= N_BONDS * (DMODEL / 4)) return;
    const int e = i >> 6, d4 = i & 63;
    const float4* ah = (const float4*)atom_h;
    const float4* bh = (const float4*)bond_h;
    unsigned* mu = (unsigned*)msg;
    const float4 s4 = ah[(size_t)src[e] * 64 + d4];
    const float4 b4 = bh[i];
    mu[(size_t)e * 256 + d4 * 2]       = pk2(s4.x + b4.x, s4.y + b4.y);
    mu[(size_t)e * 256 + d4 * 2 + 1]   = pk2(s4.z + b4.z, s4.w + b4.w);
    const float4 t4 = ah[(size_t)tgt[e] * 64 + d4];
    mu[(size_t)e * 256 + 128 + d4 * 2]     = pk2(t4.x, t4.y);
    mu[(size_t)e * 256 + 128 + d4 * 2 + 1] = pk2(t4.z, t4.w);
}

__global__ void scatter_add_edges(const float* __restrict__ msg_out,
                                  const int* __restrict__ tgt,
                                  float* __restrict__ agg)
{
    const int i = blockIdx.x * blockDim.x + threadIdx.x;
    if (i >= N_BONDS * DMODEL) return;
    const int e = i >> 8, d = i & 255;
    atomicAdd(&agg[(size_t)tgt[e] * DMODEL + d], msg_out[i]);
}

// out = LN(x [+ a] [+ b2]) * g + bb ; one wave per 256-wide row, float4 I/O.
// Optionally also writes a bf16 copy.
__global__ void add_norm(const float* __restrict__ x,
                         const float* __restrict__ a,
                         const float* __restrict__ b2,
                         const float* __restrict__ g,
                         const float* __restrict__ bb,
                         float* __restrict__ out,
                         __bf16* __restrict__ out_bf,
                         int n)
{
    const int row = blockIdx.x * 8 + (threadIdx.x >> 5);
    if (row >= n) return;
    const int lane = threadIdx.x & 31;

    const float4* xr = (const float4*)(x + (size_t)row * DMODEL);
    float4 v0 = xr[lane * 2], v1 = xr[lane * 2 + 1];
    if (a) {
        const float4* ar = (const float4*)(a + (size_t)row * DMODEL);
        const float4 a0 = ar[lane * 2], a1 = ar[lane * 2 + 1];
        v0.x += a0.x; v0.y += a0.y; v0.z += a0.z; v0.w += a0.w;
        v1.x += a1.x; v1.y += a1.y; v1.z += a1.z; v1.w += a1.w;
    }
    if (b2) {
        const float4* br = (const float4*)(b2 + (size_t)row * DMODEL);
        const float4 c0 = br[lane * 2], c1 = br[lane * 2 + 1];
        v0.x += c0.x; v0.y += c0.y; v0.z += c0.z; v0.w += c0.w;
        v1.x += c1.x; v1.y += c1.y; v1.z += c1.z; v1.w += c1.w;
    }
    float s = v0.x + v0.y + v0.z + v0.w + v1.x + v1.y + v1.z + v1.w;
    s += __shfl_xor(s, 1, 32); s += __shfl_xor(s, 2, 32);
    s += __shfl_xor(s, 4, 32); s += __shfl_xor(s, 8, 32);
    s += __shfl_xor(s, 16, 32);
    const float mean = s * (1.0f / DMODEL);
    float var = 0.f;
    {
        float d;
        d = v0.x - mean; var += d * d;  d = v0.y - mean; var += d * d;
        d = v0.z - mean; var += d * d;  d = v0.w - mean; var += d * d;
        d = v1.x - mean; var += d * d;  d = v1.y - mean; var += d * d;
        d = v1.z - mean; var += d * d;  d = v1.w - mean; var += d * d;
    }
    var += __shfl_xor(var, 1, 32); var += __shfl_xor(var, 2, 32);
    var += __shfl_xor(var, 4, 32); var += __shfl_xor(var, 8, 32);
    var += __shfl_xor(var, 16, 32);
    var *= (1.0f / DMODEL);
    const float inv = rsqrtf(var + 1e-5f);

    const float4* gr = (const float4*)g;
    const float4* br = (const float4*)bb;
    const float4 g0 = gr[lane * 2], g1 = gr[lane * 2 + 1];
    const float4 c0 = br[lane * 2], c1 = br[lane * 2 + 1];
    float4 o0, o1;
    o0.x = (v0.x - mean) * inv * g0.x + c0.x;  o0.y = (v0.y - mean) * inv * g0.y + c0.y;
    o0.z = (v0.z - mean) * inv * g0.z + c0.z;  o0.w = (v0.w - mean) * inv * g0.w + c0.w;
    o1.x = (v1.x - mean) * inv * g1.x + c1.x;  o1.y = (v1.y - mean) * inv * g1.y + c1.y;
    o1.z = (v1.z - mean) * inv * g1.z + c1.z;  o1.w = (v1.w - mean) * inv * g1.w + c1.w;
    float4* orow = (float4*)(out + (size_t)row * DMODEL);
    orow[lane * 2]     = o0;
    orow[lane * 2 + 1] = o1;
    if (out_bf) {
        unsigned* ob = (unsigned*)(out_bf + (size_t)row * DMODEL);
        ob[lane * 4]     = pk2(o0.x, o0.y);
        ob[lane * 4 + 1] = pk2(o0.z, o0.w);
        ob[lane * 4 + 2] = pk2(o1.x, o1.y);
        ob[lane * 4 + 3] = pk2(o1.z, o1.w);
    }
}

__global__ void pool_scatter(const float* __restrict__ pooled,
                             const int* __restrict__ batch_idx,
                             float* __restrict__ sums,
                             float* __restrict__ counts)
{
    const int i = blockIdx.x * blockDim.x + threadIdx.x;
    if (i >= N_ATOMS * DMODEL) return;
    const int n = i >> 8, d = i & 255;
    const int b = batch_idx[n];
    atomicAdd(&sums[(size_t)b * DMODEL + d], pooled[i]);
    if (d == 0) atomicAdd(&counts[b], 1.0f);
}

__global__ void pool_div(float* __restrict__ sums, const float* __restrict__ counts)
{
    const int i = blockIdx.x * blockDim.x + threadIdx.x;
    if (i >= NGRAPH * DMODEL) return;
    sums[i] /= fmaxf(counts[i >> 8], 1.0f);
}

// ---------------------------------------------------------------------------
// Host orchestration
// ---------------------------------------------------------------------------
extern "C" void kernel_launch(void* const* d_in, const int* in_sizes, int n_in,
                              void* d_out, int out_size, void* d_ws, size_t ws_size,
                              hipStream_t stream)
{
    const float* atom_features = (const float*)d_in[0];
    const float* bond_features = (const float*)d_in[1];
    const int*   edge_indices  = (const int*)  d_in[2];
    const int*   batch_idx     = (const int*)  d_in[3];
    const float* atom_w        = (const float*)d_in[4];
    const float* atom_b        = (const float*)d_in[5];
    const float* bond_w        = (const float*)d_in[6];
    const float* bond_b        = (const float*)d_in[7];
    const float* conv_w        = (const float*)d_in[8];
    const float* conv_b        = (const float*)d_in[9];
    const float* attn_in_w     = (const float*)d_in[10];
    const float* attn_in_b     = (const float*)d_in[11];
    const float* attn_out_w    = (const float*)d_in[12];
    const float* attn_out_b    = (const float*)d_in[13];
    const float* ln_g          = (const float*)d_in[14];
    const float* ln_b          = (const float*)d_in[15];
    const float* gattn_in_w    = (const float*)d_in[16];
    const float* gattn_in_b    = (const float*)d_in[17];
    const float* gattn_out_w   = (const float*)d_in[18];
    const float* gattn_out_b   = (const float*)d_in[19];
    const float* gn_g          = (const float*)d_in[20];
    const float* gn_b          = (const float*)d_in[21];

    const int* src = edge_indices;
    const int* tgt = edge_indices + N_BONDS;

    char* cur = (char*)d_ws;
    auto alloc_f = [&](size_t n) { float*  p = (float*) cur; cur += n * 4; return p; };
    auto alloc_b = [&](size_t n) { __bf16* p = (__bf16*)cur; cur += n * 2; return p; };

    // fp32 buffers
    float* atom_h   = alloc_f((size_t)N_ATOMS * DMODEL);
    float* bond_h   = alloc_f((size_t)N_BONDS * DMODEL);
    float* msg_out  = alloc_f((size_t)N_BONDS * DMODEL);
    float* attended = alloc_f((size_t)N_ATOMS * DMODEL);
    float* agg      = alloc_f((size_t)N_ATOMS * DMODEL);
    float* pooled   = alloc_f((size_t)N_ATOMS * DMODEL);
    float* counts   = alloc_f(NGRAPH);
    // bf16 buffers
    __bf16* atom_h_bf   = alloc_b((size_t)N_ATOMS * DMODEL);
    __bf16* messages_bf = alloc_b((size_t)N_BONDS * 2 * DMODEL);
    __bf16* qkv_bf      = alloc_b((size_t)N_ATOMS * 3 * DMODEL);
    __bf16* vT          = alloc_b((size_t)DMODEL * N_ATOMS);
    __bf16* attn_tmp_bf = alloc_b((size_t)N_ATOMS * DMODEL);
    __bf16* atom_pad    = alloc_b((size_t)N_ATOMS * 96);
    __bf16* atomw_pad   = alloc_b((size_t)DMODEL * 96);
    __bf16* bond_pad    = alloc_b((size_t)N_BONDS * 32);
    __bf16* bondw_pad   = alloc_b((size_t)DMODEL * 32);
    __bf16* convw_bf    = alloc_b((size_t)NLAYER * DMODEL * 2 * DMODEL);
    __bf16* ain_bf      = alloc_b((size_t)NLAYER * 3 * DMODEL * DMODEL);
    __bf16* aout_bf     = alloc_b((size_t)NLAYER * DMODEL * DMODEL);
    __bf16* gin_bf      = alloc_b((size_t)3 * DMODEL * DMODEL);
    __bf16* gout_bf     = alloc_b((size_t)DMODEL * DMODEL);

    float* out = (float*)d_out;

    auto cdiv = [](int a, int b) { return (a + b - 1) / b; };
    auto gemm = [&](const __bf16* A, const __bf16* W, const float* B,
                    float* Cf, __bf16* Cb, __bf16* vt,
                    int M, int N, int K, int relu) {
        const int tiles = (M / 32) * (N / 64);
        gemm_bf16_wmma<<<cdiv(tiles, 8), 256, 0, stream>>>(A, W, B, Cf, Cb, vt, M, N, K, relu);
    };
    auto cvt = [&](const float* s, __bf16* d, int n) {
        cvt_bf16<<<cdiv(n / 2, 256), 256, 0, stream>>>(s, d, n);
    };

    const int EW = N_BONDS * DMODEL;
    const int AW = N_ATOMS * DMODEL;

    // One-time conversions: pad odd-K projection operands; weights to bf16.
    pad_rows_bf<<<cdiv(N_ATOMS * 96, 256), 256, 0, stream>>>(atom_features, atom_pad, N_ATOMS, 74, 96);
    pad_rows_bf<<<cdiv(DMODEL * 96, 256), 256, 0, stream>>>(atom_w, atomw_pad, DMODEL, 74, 96);
    pad_rows_bf<<<cdiv(N_BONDS * 32, 256), 256, 0, stream>>>(bond_features, bond_pad, N_BONDS, 12, 32);
    pad_rows_bf<<<cdiv(DMODEL * 32, 256), 256, 0, stream>>>(bond_w, bondw_pad, DMODEL, 12, 32);
    cvt(conv_w,      convw_bf, NLAYER * DMODEL * 2 * DMODEL);
    cvt(attn_in_w,   ain_bf,   NLAYER * 3 * DMODEL * DMODEL);
    cvt(attn_out_w,  aout_bf,  NLAYER * DMODEL * DMODEL);
    cvt(gattn_in_w,  gin_bf,   3 * DMODEL * DMODEL);
    cvt(gattn_out_w, gout_bf,  DMODEL * DMODEL);

    // Input projections
    gemm(atom_pad, atomw_pad, atom_b, atom_h, atom_h_bf, nullptr, N_ATOMS, DMODEL, 96, 0);
    gemm(bond_pad, bondw_pad, bond_b, bond_h, nullptr, nullptr, N_BONDS, DMODEL, 32, 0);

    for (int i = 0; i < NLAYER; ++i) {
        // message passing
        build_messages<<<cdiv(EW / 4, 256), 256, 0, stream>>>(atom_h, bond_h, src, tgt, messages_bf);
        gemm(messages_bf, convw_bf + (size_t)i * DMODEL * 2 * DMODEL, conv_b + i * DMODEL,
             msg_out, nullptr, nullptr, N_BONDS, DMODEL, 2 * DMODEL, 1);
        zero_f32<<<cdiv(AW, 256), 256, 0, stream>>>(agg, AW);
        scatter_add_edges<<<cdiv(EW, 256), 256, 0, stream>>>(msg_out, tgt, agg);

        // self-attention
        gemm(atom_h_bf, ain_bf + (size_t)i * 3 * DMODEL * DMODEL, attn_in_b + i * 3 * DMODEL,
             nullptr, qkv_bf, vT, N_ATOMS, 3 * DMODEL, DMODEL, 0);
        attn_flash<<<cdiv((N_ATOMS / 16) * NHEAD, 8), 256, 0, stream>>>(
            qkv_bf, vT, batch_idx, attn_tmp_bf, N_ATOMS, 0);
        gemm(attn_tmp_bf, aout_bf + (size_t)i * DMODEL * DMODEL, attn_out_b + i * DMODEL,
             attended, nullptr, nullptr, N_ATOMS, DMODEL, DMODEL, 0);

        // residual + layernorm (f32 + bf16 copies)
        add_norm<<<cdiv(N_ATOMS, 8), 256, 0, stream>>>(
            atom_h, agg, attended, ln_g + i * DMODEL, ln_b + i * DMODEL,
            atom_h, atom_h_bf, N_ATOMS);
    }

    // graph-masked pooled attention
    gemm(atom_h_bf, gin_bf, gattn_in_b, nullptr, qkv_bf, vT, N_ATOMS, 3 * DMODEL, DMODEL, 0);
    attn_flash<<<cdiv((N_ATOMS / 16) * NHEAD, 8), 256, 0, stream>>>(
        qkv_bf, vT, batch_idx, attn_tmp_bf, N_ATOMS, 1);
    gemm(attn_tmp_bf, gout_bf, gattn_out_b, attended, nullptr, nullptr, N_ATOMS, DMODEL, DMODEL, 0);
    add_norm<<<cdiv(N_ATOMS, 8), 256, 0, stream>>>(
        attended, nullptr, nullptr, gn_g, gn_b, pooled, nullptr, N_ATOMS);

    // segment mean
    zero_f32<<<cdiv(NGRAPH * DMODEL, 256), 256, 0, stream>>>(out, NGRAPH * DMODEL);
    zero_f32<<<1, 64, 0, stream>>>(counts, NGRAPH);
    pool_scatter<<<cdiv(AW, 256), 256, 0, stream>>>(pooled, batch_idx, out, counts);
    pool_div<<<cdiv(NGRAPH * DMODEL, 256), 256, 0, stream>>>(out, counts);
}
